// VQVAEMLP_35802847379969
// MI455X (gfx1250) — matrix operand order repaired and checked by reading, not compiled, not run on hardware
//
#include <hip/hip_runtime.h>
#include <hip/hip_bf16.h>

// Fused VQ-VAE MLP for gfx1250 (MI455X), wave32 + v_wmma_f32_16x16x32_f16.
// Roofline: ~11.7 GFLOP, ~15 MB HBM traffic -> compute-bound; f16 WMMA with
// f32 accumulation is the fastest sane precision (inputs O(1), weights small).

typedef __attribute__((ext_vector_type(16))) _Float16 v16h;
typedef __attribute__((ext_vector_type(8)))  _Float16 v8h;
typedef __attribute__((ext_vector_type(8)))  float    v8f;
typedef __attribute__((ext_vector_type(4)))  float    v4f;

#define WMMA_F16(a, b, c) \
  __builtin_amdgcn_wmma_f32_16x16x32_f16(false, (a), false, (b), (short)0, (c), false, false)

namespace {
constexpr int kB      = 2048;
constexpr int kS      = 256;
constexpr int kNTok   = kB * kS;      // 524288
constexpr int kEmbed  = 4;
constexpr int kCodes  = 256;
constexpr int kRecoN  = kNTok * 3;    // 1572864 floats of x_reco
}

__global__ __launch_bounds__(256) void vqvae_fused_kernel(
    const float* __restrict__ x, const float* __restrict__ mask,
    const float* __restrict__ xcond,
    const float* __restrict__ ew1, const float* __restrict__ eb1,
    const float* __restrict__ ew2, const float* __restrict__ eb2,
    const float* __restrict__ ew3, const float* __restrict__ eb3,
    const float* __restrict__ cbk,
    const float* __restrict__ dw1, const float* __restrict__ db1,
    const float* __restrict__ dw2, const float* __restrict__ db2,
    const float* __restrict__ dw3, const float* __restrict__ db3,
    float* __restrict__ out_reco, float* __restrict__ out_loss,
    float* __restrict__ out_idx)
{
  // ---- LDS: weights pre-transposed/padded to WMMA-B geometry ([n][Kpad], f16)
  __shared__ __align__(16) _Float16 sE1[64 * 32];   // enc1: K=7 ->32, N=64
  __shared__ __align__(16) _Float16 sE2[64 * 64];   // enc2: K=64, N=64
  __shared__ __align__(16) _Float16 sE3[16 * 64];   // enc3: K=64, N=4 ->16
  __shared__ __align__(16) _Float16 sD1[64 * 32];   // dec1: K=8 ->32, N=64
  __shared__ __align__(16) _Float16 sD2[64 * 64];   // dec2
  __shared__ __align__(16) _Float16 sD3[16 * 64];   // dec3: N=3 ->16
  __shared__ __align__(16) float sBE1[64], sBE2[64], sBE3[16];
  __shared__ __align__(16) float sBD1[64], sBD2[64], sBD3[16];
  __shared__ __align__(16) float sCB[kCodes * 4];   // codebook f32
  __shared__ __align__(16) float sHP[kCodes];       // 0.5*|c|^2
  __shared__ __align__(16) _Float16 sH[8][16 * 64]; // per-wave activation tile
  __shared__ __align__(16) float sZ[8][16 * 4];     // per-wave z (f32)

  const int tid = threadIdx.x;

  // ---- Stage weights (f16, transposed, zero-padded) -------------------------
  for (int i = tid; i < 64 * 32; i += 256) {
    int n = i >> 5, k = i & 31;
    sE1[i] = (k < 7) ? (_Float16)ew1[k * 64 + n] : (_Float16)0.f;
    sD1[i] = (k < 8) ? (_Float16)dw1[k * 64 + n] : (_Float16)0.f;
  }
  for (int i = tid; i < 64 * 64; i += 256) {
    int n = i >> 6, k = i & 63;
    sE2[i] = (_Float16)ew2[k * 64 + n];
    sD2[i] = (_Float16)dw2[k * 64 + n];
  }
  for (int i = tid; i < 16 * 64; i += 256) {
    int n = i >> 6, k = i & 63;
    sE3[i] = (n < 4) ? (_Float16)ew3[k * 4 + n] : (_Float16)0.f;
    sD3[i] = (n < 3) ? (_Float16)dw3[k * 3 + n] : (_Float16)0.f;
  }
  if (tid < 64) {
    sBE1[tid] = eb1[tid]; sBE2[tid] = eb2[tid];
    sBD1[tid] = db1[tid]; sBD2[tid] = db2[tid];
  }
  if (tid < 16) {
    sBE3[tid] = (tid < 4) ? eb3[tid] : 0.f;
    sBD3[tid] = (tid < 3) ? db3[tid] : 0.f;
  }
  for (int i = tid; i < kCodes * 4; i += 256) sCB[i] = cbk[i];
  for (int c = tid; c < kCodes; c += 256) {
    float c0 = cbk[c * 4 + 0], c1 = cbk[c * 4 + 1];
    float c2 = cbk[c * 4 + 2], c3 = cbk[c * 4 + 3];
    sHP[c] = 0.5f * (c0 * c0 + c1 * c1 + c2 * c2 + c3 * c3);
  }
  __syncthreads();

  const int lane = tid & 31;
  const int wv   = tid >> 5;
  const int lo16 = lane & 15;
  const bool hi  = (lane & 16) != 0;
  const int tile = blockIdx.x * 128 + wv * 16;   // 16 tokens per wave
  _Float16* hb = sH[wv];
  float*    zb = sZ[wv];

  // B fragment: lane n = nt*16+lo16, K = ks*32 + (hi?16:0) .. +15, contiguous.
  auto ldB = [&](const _Float16* w, int Kpad, int nt, int ks) -> v16h {
    int n  = nt * 16 + lo16;
    int kb = ks * 32 + (hi ? 16 : 0);
    const v8h* p = (const v8h*)(w + n * Kpad + kb);
    v8h b0 = p[0], b1 = p[1];
    return __builtin_shufflevector(b0, b1, 0, 1, 2, 3, 4, 5, 6, 7,
                                           8, 9, 10, 11, 12, 13, 14, 15);
  };
  // A fragment from activation tile [16][64] f16 (row = lo16).
  auto ldA = [&](int ks) -> v16h {
    int kb = ks * 32 + (hi ? 16 : 0);
    const _Float16* base = hb + lo16 * 64 + kb;
    v8h a0 = *(const v8h*)(base);        // K kb..kb+7
    v8h a1 = *(const v8h*)(base + 8);    // K kb+8..kb+15
    return __builtin_shufflevector(a0, a1, 0, 1, 2, 3, 4, 5, 6, 7,
                                           8, 9, 10, 11, 12, 13, 14, 15);
  };
  // Fold bias into C initializer (per-column broadcast).
  auto biasC = [&](const float* bias, int nt) -> v8f {
    float bb = bias[nt * 16 + lo16];
    v8f c;
#pragma unroll
    for (int i = 0; i < 8; ++i) c[i] = bb;
    return c;
  };
  // Store C tile into activation tile with ReLU + f16 convert.
  auto stC = [&](const v8f& c, int nt) {
    int col = nt * 16 + lo16;
    int rb  = hi ? 8 : 0;
#pragma unroll
    for (int r = 0; r < 8; ++r) {
      float v = c[r];
      v = v > 0.f ? v : 0.f;
      hb[(rb + r) * 64 + col] = (_Float16)v;
    }
  };

  // ---- Per-token inputs (lane lo16 owns token t) ----------------------------
  const int t    = tile + lo16;
  const int brow = t >> 8;                     // S = 256
  const float m  = mask[t];
  const v4f cnd  = *(const v4f*)(xcond + brow * 4);
  const float x0 = x[t * 3 + 0], x1 = x[t * 3 + 1], x2 = x[t * 3 + 2];

  // ---- Encoder layer 1: [x,cond]*mask (K=7 padded to 32) --------------------
  v16h aIn;
#pragma unroll
  for (int i = 0; i < 16; ++i) aIn[i] = (_Float16)0.f;
  if (!hi) {
    aIn[0] = (_Float16)(x0 * m);   aIn[1] = (_Float16)(x1 * m);
    aIn[2] = (_Float16)(x2 * m);   aIn[3] = (_Float16)(cnd.x * m);
    aIn[4] = (_Float16)(cnd.y * m); aIn[5] = (_Float16)(cnd.z * m);
    aIn[6] = (_Float16)(cnd.w * m);
  }
  {
    v8f h1[4];
#pragma unroll
    for (int nt = 0; nt < 4; ++nt) {
      v8f c = biasC(sBE1, nt);
      h1[nt] = WMMA_F16(aIn, ldB(sE1, 32, nt, 0), c);
    }
#pragma unroll
    for (int nt = 0; nt < 4; ++nt) stC(h1[nt], nt);
  }
  __syncthreads();

  // ---- Encoder layer 2: 64 -> 64 -------------------------------------------
  v16h a0 = ldA(0), a1 = ldA(1);
  __syncthreads();
  {
    v8f h2[4];
#pragma unroll
    for (int nt = 0; nt < 4; ++nt) {
      v8f c = biasC(sBE2, nt);
      c = WMMA_F16(a0, ldB(sE2, 64, nt, 0), c);
      c = WMMA_F16(a1, ldB(sE2, 64, nt, 1), c);
      h2[nt] = c;
    }
#pragma unroll
    for (int nt = 0; nt < 4; ++nt) stC(h2[nt], nt);
  }
  __syncthreads();

  // ---- Encoder layer 3: 64 -> 4 (N padded to 16, no ReLU) -------------------
  a0 = ldA(0); a1 = ldA(1);
  {
    v8f zc = biasC(sBE3, 0);
    zc = WMMA_F16(a0, ldB(sE3, 64, 0, 0), zc);
    zc = WMMA_F16(a1, ldB(sE3, 64, 0, 1), zc);
    if (lo16 < 4) {
      int rb = hi ? 8 : 0;
#pragma unroll
      for (int r = 0; r < 8; ++r) zb[(rb + r) * 4 + lo16] = zc[r];
    }
  }
  __syncthreads();

  // ---- VQ: argmin(0.5|c|^2 - z.c); halves of wave split the 256 codes -------
  const v4f z = *(const v4f*)(zb + lo16 * 4);
  float bd = 3.4e38f;
  int   bi = 0;
  const int cb0 = hi ? 128 : 0;
#pragma unroll 4
  for (int j = 0; j < 128; ++j) {
    int ci = cb0 + j;
    v4f cv = *(const v4f*)(sCB + ci * 4);
    float dot = z.x * cv.x + z.y * cv.y + z.z * cv.z + z.w * cv.w;
    float d = sHP[ci] - dot;
    if (d < bd) { bd = d; bi = ci; }
  }
  {
    float od = __shfl_xor(bd, 16, 32);
    int   oi = __shfl_xor(bi, 16, 32);
    if (od < bd || (od == bd && oi < bi)) { bd = od; bi = oi; }
  }
  const v4f q = *(const v4f*)(sCB + bi * 4);

  float ls = 0.f;
  if (!hi) {
    out_idx[t] = (float)bi;
    float d0 = z.x - q.x, d1 = z.y - q.y, d2 = z.z - q.z, d3 = z.w - q.w;
    ls = d0 * d0 + d1 * d1 + d2 * d2 + d3 * d3;
  }
#pragma unroll
  for (int o = 16; o >= 1; o >>= 1) ls += __shfl_xor(ls, o, 32);
  if (lane == 0)
    atomicAdd(out_loss, ls * (1.25f / (float)(kNTok * kEmbed)));

  // ---- Decoder layer 1: [z_q,cond]*mask (K=8 padded to 32) ------------------
  v16h aD;
#pragma unroll
  for (int i = 0; i < 16; ++i) aD[i] = (_Float16)0.f;
  if (!hi) {
    aD[0] = (_Float16)(q.x * m);   aD[1] = (_Float16)(q.y * m);
    aD[2] = (_Float16)(q.z * m);   aD[3] = (_Float16)(q.w * m);
    aD[4] = (_Float16)(cnd.x * m); aD[5] = (_Float16)(cnd.y * m);
    aD[6] = (_Float16)(cnd.z * m); aD[7] = (_Float16)(cnd.w * m);
  }
  {
    v8f g1[4];
#pragma unroll
    for (int nt = 0; nt < 4; ++nt) {
      v8f c = biasC(sBD1, nt);
      g1[nt] = WMMA_F16(aD, ldB(sD1, 32, nt, 0), c);
    }
#pragma unroll
    for (int nt = 0; nt < 4; ++nt) stC(g1[nt], nt);
  }
  __syncthreads();

  // ---- Decoder layer 2: 64 -> 64 -------------------------------------------
  a0 = ldA(0); a1 = ldA(1);
  __syncthreads();
  {
    v8f g2[4];
#pragma unroll
    for (int nt = 0; nt < 4; ++nt) {
      v8f c = biasC(sBD2, nt);
      c = WMMA_F16(a0, ldB(sD2, 64, nt, 0), c);
      c = WMMA_F16(a1, ldB(sD2, 64, nt, 1), c);
      g2[nt] = c;
    }
#pragma unroll
    for (int nt = 0; nt < 4; ++nt) stC(g2[nt], nt);
  }
  __syncthreads();

  // ---- Decoder layer 3: 64 -> 3 (N padded to 16, no ReLU) -------------------
  a0 = ldA(0); a1 = ldA(1);
  {
    v8f oc = biasC(sBD3, 0);
    oc = WMMA_F16(a0, ldB(sD3, 64, 0, 0), oc);
    oc = WMMA_F16(a1, ldB(sD3, 64, 0, 1), oc);
    if (lo16 < 3) {
      int rb = hi ? 8 : 0;
#pragma unroll
      for (int r = 0; r < 8; ++r) {
        int row = tile + rb + r;
        out_reco[row * 3 + lo16] = oc[r];
      }
    }
  }
}

extern "C" void kernel_launch(void* const* d_in, const int* in_sizes, int n_in,
                              void* d_out, int out_size, void* d_ws, size_t ws_size,
                              hipStream_t stream) {
  (void)in_sizes; (void)n_in; (void)out_size; (void)d_ws; (void)ws_size;
  const float* x     = (const float*)d_in[0];
  const float* mask  = (const float*)d_in[1];
  const float* xcond = (const float*)d_in[2];
  const float* ew1   = (const float*)d_in[3];
  const float* eb1   = (const float*)d_in[4];
  const float* ew2   = (const float*)d_in[5];
  const float* eb2   = (const float*)d_in[6];
  const float* ew3   = (const float*)d_in[7];
  const float* eb3   = (const float*)d_in[8];
  const float* cbk   = (const float*)d_in[9];
  const float* dw1   = (const float*)d_in[10];
  const float* db1   = (const float*)d_in[11];
  const float* dw2   = (const float*)d_in[12];
  const float* db2   = (const float*)d_in[13];
  const float* dw3   = (const float*)d_in[14];
  const float* db3   = (const float*)d_in[15];

  float* out      = (float*)d_out;
  float* out_reco = out;                 // [2048*256*3]
  float* out_loss = out + kRecoN;        // [1]
  float* out_idx  = out + kRecoN + 1;    // [2048*256] stored as float

  // Loss is accumulated with atomics; zero it every launch (graph-capture OK).
  hipMemsetAsync(out_loss, 0, sizeof(float), stream);

  const int blocks = kNTok / 128;        // 16 tokens/wave * 8 waves/block
  vqvae_fused_kernel<<<blocks, 256, 0, stream>>>(
      x, mask, xcond, ew1, eb1, ew2, eb2, ew3, eb3, cbk,
      dw1, db1, dw2, db2, dw3, db3, out_reco, out_loss, out_idx);
}
